// Transformer_76974403879493
// MI455X (gfx1250) — matrix operand order repaired and checked by reading, not compiled, not run on hardware
//
#include <hip/hip_runtime.h>
#include <hip/hip_bf16.h>

// ---------------------------------------------------------------------------
// Problem constants (from the reference):
//   L=6, B=4, T=1024, D=512, H=8, F=2048, DH=64, BIAS_DIM=2, E=65536
// ---------------------------------------------------------------------------
#define LNUM 6
#define BSZ  4
#define TSEQ 1024
#define DMOD 512
#define HNUM 8
#define FDIM 2048
#define DHD  64
#define EDIM 65536
#define BTROWS (BSZ * TSEQ)   // 4096
#define LN_EPS 1.0e-3f

typedef __attribute__((ext_vector_type(16))) _Float16 v16h;
typedef __attribute__((ext_vector_type(8)))  _Float16 v8h;
typedef __attribute__((ext_vector_type(8)))  float    v8f;

// ---------------------------------------------------------------------------
// WMMA helpers (gfx1250, wave32). Fragment layouts per CDNA5 ISA.
// ---------------------------------------------------------------------------
__device__ __forceinline__ v8f wmma_f32_f16(v16h a, v16h b, v8f c) {
  // 8 args: (neg_a, A, neg_b, B, c_mod, C, reuse_a, reuse_b)
  return __builtin_amdgcn_wmma_f32_16x16x32_f16(false, a, false, b, (short)0, c,
                                                false, false);
}

// A-matrix 16x32 f16 fragment from a row-major source (generic pointer: global
// or LDS). Element (m,k) at A[m*lda + k].
// Lane L: m = L&15; VGPR0-3 hold k = kb..kb+7 (kb = (L>=16)?8:0),
// VGPR4-7 hold k = 16+kb .. 16+kb+7.  -> two 16B vector loads per lane.
__device__ __forceinline__ v16h load_a_frag(const _Float16* A, int lda) {
  int lane = threadIdx.x & 31;
  int m  = lane & 15;
  int kb = (lane >> 4) << 3;           // 0 or 8
  const _Float16* p = A + (size_t)m * lda + kb;
  v8h lo = *(const v8h*)(p);
  v8h hi = *(const v8h*)(p + 16);
  v16h f;
#pragma unroll
  for (int i = 0; i < 8; ++i) { f[i] = lo[i]; f[i + 8] = hi[i]; }
  return f;
}

// B-matrix 32x16 f16 fragment where element (k,n) sits at B[n*ldn + k]
// (k contiguous -> two 16B vector loads per lane).
// Lane L: n = L&15; VGPR0-3 hold k = kb..kb+7 (kb = (L>=16)?8:0), VGPR4-7 hold k = 16+kb..16+kb+7.
__device__ __forceinline__ v16h load_b_frag_kcontig(const _Float16* B, int ldn) {
  int lane = threadIdx.x & 31;
  int n  = lane & 15;
  int kb = (lane >> 4) << 3;           // 0 or 8
  const _Float16* p = B + (size_t)n * ldn + kb;
  v8h lo = *(const v8h*)(p);
  v8h hi = *(const v8h*)(p + 16);
  v16h f;
#pragma unroll
  for (int i = 0; i < 8; ++i) { f[i] = lo[i]; f[i + 8] = hi[i]; }
  return f;
}

// ---------------------------------------------------------------------------
// Weight convert + transpose: in fp32 [l][K][N] row-major -> out f16 [l][N][K]
// (k-contiguous so GEMM B-fragments are two b128 loads per lane).
// ---------------------------------------------------------------------------
__global__ void transpose_w_f16_kernel(const float* __restrict__ in,
                                       _Float16* __restrict__ out,
                                       int Kd, int Nd, long total) {
  long idx = (long)blockIdx.x * blockDim.x + threadIdx.x;
  if (idx >= total) return;
  long per = (long)Kd * Nd;
  long l = idx / per;
  long w = idx - l * per;
  int k = (int)(w / Nd);
  int n = (int)(w - (long)k * Nd);
  out[l * per + (long)n * Kd + k] = (_Float16)in[idx];
}

// ---------------------------------------------------------------------------
// LayerNorm over D=512. One block (256 threads) per (b,t) row.
// ---------------------------------------------------------------------------
__global__ void ln_kernel(const float* __restrict__ x,
                          const float* __restrict__ sc,
                          const float* __restrict__ bi,
                          _Float16* __restrict__ oh, float* __restrict__ of) {
  __shared__ float red[256];
  int row = blockIdx.x;
  int tid = threadIdx.x;
  const float* xr = x + (size_t)row * DMOD;
  float a0 = xr[tid], a1 = xr[tid + 256];
  red[tid] = a0 + a1;
  __syncthreads();
  for (int s = 128; s > 0; s >>= 1) {
    if (tid < s) red[tid] += red[tid + s];
    __syncthreads();
  }
  float mean = red[0] * (1.0f / (float)DMOD);
  __syncthreads();
  float d0 = a0 - mean, d1 = a1 - mean;
  red[tid] = d0 * d0 + d1 * d1;
  __syncthreads();
  for (int s = 128; s > 0; s >>= 1) {
    if (tid < s) red[tid] += red[tid + s];
    __syncthreads();
  }
  float rinv = rsqrtf(red[0] * (1.0f / (float)DMOD) + LN_EPS);
  float y0 = d0 * rinv * sc[tid] + bi[tid];
  float y1 = d1 * rinv * sc[tid + 256] + bi[tid + 256];
  size_t base = (size_t)row * DMOD;
  if (oh) {
    oh[base + tid]       = (_Float16)y0;
    oh[base + tid + 256] = (_Float16)y1;
  } else {
    of[base + tid]       = y0;
    of[base + tid + 256] = y1;
  }
}

// ---------------------------------------------------------------------------
// Blocked WMMA GEMM: C[M,N] = A_h[M,K] x Bt_h[N,K]^T (+bias)(+relu)
// Block = 128 threads = 4 waves covering 64(M) x 64(N). Each wave owns one
// 16-col N-slab and 4 M-subtiles -> 4 WMMAs per B fragment (B reuse x4).
// The shared 64x32 A tile is staged into LDS with CDNA5 async-to-LDS loads
// (ASYNCcnt), then each wave reads its A fragments as contiguous ds_b128.
// flags: 1 = accumulate into Cf (residual); 2 = relu; 4 = f16 out row-major;
//        8 = f16 out in attention-V layout (B,H,DH,T)  [assumes M=B*T, N=D].
// ---------------------------------------------------------------------------
__global__ void gemm_wmma_kernel(const _Float16* __restrict__ A,
                                 const _Float16* __restrict__ Bt,
                                 const float* __restrict__ bias,
                                 float* __restrict__ Cf,
                                 _Float16* __restrict__ Ch,
                                 int M, int N, int K, int flags) {
  __shared__ __align__(16) _Float16 aShm[64 * 32];
  int tid  = threadIdx.x;
  int wave = tid >> 5;
  int lane = tid & 31;
  int tilesN = N >> 6;
  int bm = blockIdx.x / tilesN;
  int bn = blockIdx.x - bm * tilesN;
  int m0 = bm << 6;
  int n0 = (bn << 6) + (wave << 4);
  const _Float16* Bbase = Bt + (size_t)n0 * K;   // 16 weight rows (cols of W)
  int lo = lane & 15, hw = lane >> 4;
  unsigned aBase = (unsigned)(uintptr_t)&aShm[0];  // LDS offset (low 32 bits)
  unsigned lb = aBase + (unsigned)tid * 32u;
  v8f acc0 = {}, acc1 = {}, acc2 = {}, acc3 = {};
  for (int k0 = 0; k0 < K; k0 += 32) {
    // --- async-stage A tile: 64 rows x 32 halves (4KB), 32B per thread ---
    const _Float16* gp = A + (size_t)(m0 + (tid >> 1)) * K + k0 + ((tid & 1) << 4);
    asm volatile("global_load_async_to_lds_b128 %0, %1, off"
                 :: "v"(lb), "v"(gp) : "memory");
    asm volatile("global_load_async_to_lds_b128 %0, %1, off"
                 :: "v"(lb + 16u), "v"(gp + 8) : "memory");
    if (k0 + 32 < K)
      __builtin_prefetch(Bbase + (size_t)lo * K + k0 + 32, 0, 1);
    asm volatile("s_wait_asynccnt 0x0" ::: "memory");
    __syncthreads();
    // --- fragments + 4 WMMAs ---
    v16h bf  = load_b_frag_kcontig(Bbase + k0, K);
    v16h af0 = load_a_frag(&aShm[0 * 16 * 32], 32);
    v16h af1 = load_a_frag(&aShm[1 * 16 * 32], 32);
    v16h af2 = load_a_frag(&aShm[2 * 16 * 32], 32);
    v16h af3 = load_a_frag(&aShm[3 * 16 * 32], 32);
    acc0 = wmma_f32_f16(af0, bf, acc0);
    acc1 = wmma_f32_f16(af1, bf, acc1);
    acc2 = wmma_f32_f16(af2, bf, acc2);
    acc3 = wmma_f32_f16(af3, bf, acc3);
    __syncthreads();
  }
  // --- epilogue ---
  float bcol = bias ? bias[n0 + lo] : 0.0f;
#pragma unroll
  for (int mt = 0; mt < 4; ++mt) {
    v8f acc = (mt == 0) ? acc0 : (mt == 1) ? acc1 : (mt == 2) ? acc2 : acc3;
#pragma unroll
    for (int r = 0; r < 8; ++r) {
      int row = m0 + mt * 16 + r + hw * 8;   // C/D layout: VGPR r -> M=r(+8)
      int col = n0 + lo;
      float v = acc[r] + bcol;
      if (flags & 2) v = fmaxf(v, 0.0f);
      if (flags & 8) {
        // V layout (B,H,DH,T): b = row/T, t = row%T, h = col/DH, a = col%DH
        int b = row >> 10, t = row & (TSEQ - 1);
        int h = col >> 6,  a = col & (DHD - 1);
        Ch[((((size_t)b * HNUM + h) * DHD + a) << 10) + t] = (_Float16)v;
      } else if (flags & 4) {
        Ch[(size_t)row * N + col] = (_Float16)v;
      } else if (flags & 1) {
        Cf[(size_t)row * N + col] += v;
      } else {
        Cf[(size_t)row * N + col] = v;
      }
    }
  }
}

// ---------------------------------------------------------------------------
// ksum[b,t,h] = sum_a k[b,t,h,a]
// ---------------------------------------------------------------------------
__global__ void ksum_kernel(const _Float16* __restrict__ kh,
                            float* __restrict__ ks) {
  int i = blockIdx.x * blockDim.x + threadIdx.x;   // over B*T*H
  if (i >= BTROWS * HNUM) return;
  int h = i & (HNUM - 1);
  int bt = i / HNUM;
  const _Float16* p = kh + (size_t)bt * DMOD + h * DHD;
  float s = 0.0f;
#pragma unroll 8
  for (int a = 0; a < DHD; ++a) s += (float)p[a];
  ks[i] = s;
}

// ---------------------------------------------------------------------------
// bias3[b, ki, qi] += dot(Eb[et], Es)  (E = 65536 scattered entries)
// ---------------------------------------------------------------------------
__global__ void scatter_bias_kernel(const int* __restrict__ ab,
                                    const float* __restrict__ Ebl,
                                    const float* __restrict__ Esl,
                                    float* __restrict__ bias3) {
  int e = blockIdx.x * blockDim.x + threadIdx.x;
  if (e >= EDIM) return;
  int et = ab[e * 4 + 0];
  int bi = ab[e * 4 + 1];
  int ki = ab[e * 4 + 2];
  int qi = ab[e * 4 + 3];
  const float* eb = Ebl + (size_t)et * DHD;
  float s = 0.0f;
#pragma unroll 8
  for (int a = 0; a < DHD; ++a) s += eb[a] * Esl[a];
  atomicAdd(&bias3[((size_t)bi * TSEQ + ki) * TSEQ + qi], s);
}

// ---------------------------------------------------------------------------
// Flash attention: one wave per (b, h, 16-row q-tile); online softmax over
// 32-key chunks with causal early-exit. Q*K^T and P*V via WMMA. V comes in
// pre-transposed (B,H,DH,T) so every V fragment is two b128 loads per lane.
// Bias term bias3[b,q,k]*ksum[b,k,h] folded in before scale/mask (ref order).
// ---------------------------------------------------------------------------
__global__ void flash_attn_kernel(const _Float16* __restrict__ qh,
                                  const _Float16* __restrict__ kh,
                                  const _Float16* __restrict__ vt,
                                  const float* __restrict__ bias3,
                                  const float* __restrict__ ksum,
                                  _Float16* __restrict__ ctxh) {
  __shared__ __align__(16) _Float16 pShm[16 * 32];  // P tile C->A layout bounce
  const int QT = TSEQ / 16;
  int blk = blockIdx.x;
  int qt = blk % QT;
  int h  = (blk / QT) % HNUM;
  int b  = blk / (QT * HNUM);
  int q0 = qt << 4;
  int lane = threadIdx.x;
  int lo = lane & 15, hw = lane >> 4;
  const float scale = 0.125f;                       // DH^-0.5 = 1/8

  const _Float16* qbase = qh + ((size_t)(b * TSEQ + q0)) * DMOD + (size_t)h * DHD;
  v16h qa0 = load_a_frag(qbase, DMOD);              // a = 0..31
  v16h qa1 = load_a_frag(qbase + 32, DMOD);         // a = 32..63
  const _Float16* vtbase = vt + ((size_t)(b * HNUM + h) * DHD) * TSEQ;

  v8f c0 = {}, c1 = {}, c2 = {}, c3 = {};           // ctx 16 x 64 accumulator
  float mrow[8], lrow[8];
#pragma unroll
  for (int r = 0; r < 8; ++r) { mrow[r] = -3.0e38f; lrow[r] = 0.0f; }

  const float* b3row = bias3 + (size_t)b * TSEQ * TSEQ;
  int nchunks = (q0 + 16 + 31) >> 5;                // causal: skip keys > q0+15
  for (int c = 0; c < nchunks; ++c) {
    int kc = c << 5;
    // S = Q * K^T over 32 keys; B element (a, key) at K[key*DMOD + a].
    const _Float16* kbase = kh + ((size_t)(b * TSEQ + kc)) * DMOD + (size_t)h * DHD;
    v16h k00 = load_b_frag_kcontig(kbase, DMOD);
    v16h k01 = load_b_frag_kcontig(kbase + 32, DMOD);
    v16h k10 = load_b_frag_kcontig(kbase + (size_t)16 * DMOD, DMOD);
    v16h k11 = load_b_frag_kcontig(kbase + (size_t)16 * DMOD + 32, DMOD);
    v8f s0 = {}, s1 = {};
    s0 = wmma_f32_f16(qa0, k00, s0);
    s0 = wmma_f32_f16(qa1, k01, s0);
    s1 = wmma_f32_f16(qa0, k10, s1);
    s1 = wmma_f32_f16(qa1, k11, s1);

    float p0[8], p1[8];
#pragma unroll
    for (int r = 0; r < 8; ++r) {
      int q = q0 + r + hw * 8;
      int key0 = kc + lo, key1 = key0 + 16;
      float v0 = (s0[r] + b3row[(size_t)q * TSEQ + key0] *
                              ksum[(size_t)(b * TSEQ + key0) * HNUM + h]) * scale;
      float v1 = (s1[r] + b3row[(size_t)q * TSEQ + key1] *
                              ksum[(size_t)(b * TSEQ + key1) * HNUM + h]) * scale;
      v0 = (key0 <= q) ? v0 : -3.0e38f;
      v1 = (key1 <= q) ? v1 : -3.0e38f;
      float cm = fmaxf(v0, v1);
#pragma unroll
      for (int m = 1; m < 16; m <<= 1) cm = fmaxf(cm, __shfl_xor(cm, m, 32));
      float mnew = fmaxf(mrow[r], cm);
      float e0 = (key0 <= q) ? __expf(v0 - mnew) : 0.0f;
      float e1 = (key1 <= q) ? __expf(v1 - mnew) : 0.0f;
      float rs = e0 + e1;
#pragma unroll
      for (int m = 1; m < 16; m <<= 1) rs += __shfl_xor(rs, m, 32);
      float alpha = __expf(mrow[r] - mnew);
      lrow[r] = lrow[r] * alpha + rs;
      mrow[r] = mnew;
      c0[r] *= alpha; c1[r] *= alpha; c2[r] *= alpha; c3[r] *= alpha;
      p0[r] = e0; p1[r] = e1;
    }
    __syncthreads();
    // Bounce P from C-layout to A-layout through LDS.
#pragma unroll
    for (int r = 0; r < 8; ++r) {
      pShm[(r + hw * 8) * 32 + lo]      = (_Float16)p0[r];
      pShm[(r + hw * 8) * 32 + lo + 16] = (_Float16)p1[r];
    }
    __syncthreads();
    v16h pf  = load_a_frag(pShm, 32);
    // V fragments: element (key j, a) at vtbase[(ta+a)*TSEQ + kc + j].
    v16h vf0 = load_b_frag_kcontig(vtbase + (size_t)0  * TSEQ + kc, TSEQ);
    v16h vf1 = load_b_frag_kcontig(vtbase + (size_t)16 * TSEQ + kc, TSEQ);
    v16h vf2 = load_b_frag_kcontig(vtbase + (size_t)32 * TSEQ + kc, TSEQ);
    v16h vf3 = load_b_frag_kcontig(vtbase + (size_t)48 * TSEQ + kc, TSEQ);
    c0 = wmma_f32_f16(pf, vf0, c0);
    c1 = wmma_f32_f16(pf, vf1, c1);
    c2 = wmma_f32_f16(pf, vf2, c2);
    c3 = wmma_f32_f16(pf, vf3, c3);
    __syncthreads();
  }
#pragma unroll
  for (int r = 0; r < 8; ++r) {
    int row = q0 + r + hw * 8;
    float inv = 1.0f / lrow[r];
    size_t base = ((size_t)(b * TSEQ + row)) * DMOD + (size_t)h * DHD;
    ctxh[base + 0  + lo] = (_Float16)(c0[r] * inv);
    ctxh[base + 16 + lo] = (_Float16)(c1[r] * inv);
    ctxh[base + 32 + lo] = (_Float16)(c2[r] * inv);
    ctxh[base + 48 + lo] = (_Float16)(c3[r] * inv);
  }
}

// ---------------------------------------------------------------------------
// Host orchestration. Workspace partition (~97 MB).
// ---------------------------------------------------------------------------
extern "C" void kernel_launch(void* const* d_in, const int* in_sizes, int n_in,
                              void* d_out, int out_size, void* d_ws,
                              size_t ws_size, hipStream_t stream) {
  (void)in_sizes; (void)n_in; (void)out_size; (void)ws_size;
  const float* states = (const float*)d_in[0];
  /* d_in[1] = masks: causal, implicit */
  const int*   abias  = (const int*)d_in[2];
  const float* Wq = (const float*)d_in[3];
  const float* Wk = (const float*)d_in[4];
  const float* Wv = (const float*)d_in[5];
  const float* Wo = (const float*)d_in[6];
  const float* Eb = (const float*)d_in[7];
  const float* Es = (const float*)d_in[8];
  const float* ln1_s = (const float*)d_in[9];
  const float* ln1_b = (const float*)d_in[10];
  const float* ln2_s = (const float*)d_in[11];
  const float* ln2_b = (const float*)d_in[12];
  const float* lno_s = (const float*)d_in[13];
  const float* lno_b = (const float*)d_in[14];
  const float* W1 = (const float*)d_in[15];
  const float* b1 = (const float*)d_in[16];
  const float* W2 = (const float*)d_in[17];
  const float* b2 = (const float*)d_in[18];

  char* ws = (char*)d_ws;
  size_t off = 0;
  auto alloc = [&](size_t bytes) -> char* {
    char* p = ws + off;
    off += (bytes + 255) & ~(size_t)255;
    return p;
  };
  const int M = BTROWS;                                   // 4096
  float*    statesBuf = (float*)   alloc((size_t)M * DMOD * 4);
  _Float16* x_h       = (_Float16*)alloc((size_t)M * DMOD * 2);
  _Float16* q_h       = (_Float16*)alloc((size_t)M * DMOD * 2);
  _Float16* k_h       = (_Float16*)alloc((size_t)M * DMOD * 2);
  _Float16* vt_h      = (_Float16*)alloc((size_t)M * DMOD * 2);  // (B,H,DH,T)
  _Float16* ctx_h     = (_Float16*)alloc((size_t)M * DMOD * 2);
  _Float16* h_h       = (_Float16*)alloc((size_t)M * FDIM * 2);
  float*    ksumBuf   = (float*)   alloc((size_t)M * HNUM * 4);
  float*    bias3     = (float*)   alloc((size_t)BSZ * TSEQ * TSEQ * 4);
  _Float16* Wq_t = (_Float16*)alloc((size_t)LNUM * DMOD * DMOD * 2);
  _Float16* Wk_t = (_Float16*)alloc((size_t)LNUM * DMOD * DMOD * 2);
  _Float16* Wv_t = (_Float16*)alloc((size_t)LNUM * DMOD * DMOD * 2);
  _Float16* Wo_t = (_Float16*)alloc((size_t)LNUM * DMOD * DMOD * 2);
  _Float16* W1_t = (_Float16*)alloc((size_t)LNUM * DMOD * FDIM * 2);
  _Float16* W2_t = (_Float16*)alloc((size_t)LNUM * FDIM * DMOD * 2);

  // Weight convert + transpose to [l][N][K] f16 (deterministic; every call).
  const long nWdd = (long)LNUM * DMOD * DMOD;
  const long nWdf = (long)LNUM * DMOD * FDIM;
  transpose_w_f16_kernel<<<(int)((nWdd + 255) / 256), 256, 0, stream>>>(
      Wq, Wq_t, DMOD, DMOD, nWdd);
  transpose_w_f16_kernel<<<(int)((nWdd + 255) / 256), 256, 0, stream>>>(
      Wk, Wk_t, DMOD, DMOD, nWdd);
  transpose_w_f16_kernel<<<(int)((nWdd + 255) / 256), 256, 0, stream>>>(
      Wv, Wv_t, DMOD, DMOD, nWdd);
  transpose_w_f16_kernel<<<(int)((nWdd + 255) / 256), 256, 0, stream>>>(
      Wo, Wo_t, DMOD, DMOD, nWdd);
  transpose_w_f16_kernel<<<(int)((nWdf + 255) / 256), 256, 0, stream>>>(
      W1, W1_t, DMOD, FDIM, nWdf);
  transpose_w_f16_kernel<<<(int)((nWdf + 255) / 256), 256, 0, stream>>>(
      W2, W2_t, FDIM, DMOD, nWdf);

  hipMemcpyAsync(statesBuf, states, (size_t)M * DMOD * 4,
                 hipMemcpyDeviceToDevice, stream);

  for (int l = 0; l < LNUM; ++l) {
    const size_t wdd = (size_t)l * DMOD * DMOD;
    const size_t wdf = (size_t)l * DMOD * FDIM;
    // LN1 -> f16
    ln_kernel<<<M, 256, 0, stream>>>(statesBuf, ln1_s + l * DMOD,
                                     ln1_b + l * DMOD, x_h, nullptr);
    // QKV projections (f16 out; V written transposed for attention)
    int gDD = (M / 64) * (DMOD / 64);
    gemm_wmma_kernel<<<gDD, 128, 0, stream>>>(x_h, Wq_t + wdd, nullptr,
                                              nullptr, q_h, M, DMOD, DMOD, 4);
    gemm_wmma_kernel<<<gDD, 128, 0, stream>>>(x_h, Wk_t + wdd, nullptr,
                                              nullptr, k_h, M, DMOD, DMOD, 4);
    gemm_wmma_kernel<<<gDD, 128, 0, stream>>>(x_h, Wv_t + wdd, nullptr,
                                              nullptr, vt_h, M, DMOD, DMOD, 8);
    // ksum + sparse attention bias
    ksum_kernel<<<(M * HNUM + 255) / 256, 256, 0, stream>>>(k_h, ksumBuf);
    hipMemsetAsync(bias3, 0, (size_t)BSZ * TSEQ * TSEQ * 4, stream);
    scatter_bias_kernel<<<(EDIM + 255) / 256, 256, 0, stream>>>(
        abias, Eb + (size_t)l * 2 * DHD, Es + (size_t)l * DHD, bias3);
    // Fused flash attention
    flash_attn_kernel<<<BSZ * HNUM * (TSEQ / 16), 32, 0, stream>>>(
        q_h, k_h, vt_h, bias3, ksumBuf, ctx_h);
    // Output projection, residual accumulate
    gemm_wmma_kernel<<<gDD, 128, 0, stream>>>(ctx_h, Wo_t + wdd, nullptr,
                                              statesBuf, nullptr, M, DMOD,
                                              DMOD, 1);
    // LN2 -> f16, FFN1 (bias+relu, f16 out), FFN2 (bias, residual accumulate)
    ln_kernel<<<M, 256, 0, stream>>>(statesBuf, ln2_s + l * DMOD,
                                     ln2_b + l * DMOD, x_h, nullptr);
    gemm_wmma_kernel<<<(M / 64) * (FDIM / 64), 128, 0, stream>>>(
        x_h, W1_t + wdf, b1 + (size_t)l * FDIM, nullptr, h_h, M, FDIM, DMOD,
        4 | 2);
    gemm_wmma_kernel<<<gDD, 128, 0, stream>>>(
        h_h, W2_t + wdf, b2 + (size_t)l * DMOD, statesBuf, nullptr, M, DMOD,
        FDIM, 1);
  }
  // Final LN -> fp32 output
  ln_kernel<<<M, 256, 0, stream>>>(statesBuf, lno_s, lno_b, nullptr,
                                   (float*)d_out);
}
